// Bi_Dir_FeatureExtractor_57303453663963
// MI455X (gfx1250) — matrix operand order repaired and checked by reading, not compiled, not run on hardware
//
#include <hip/hip_runtime.h>
#include <hip/hip_bf16.h>

// ---------------------------------------------------------------------------
// Bi-directional feature extractor for MI455X (gfx1250, wave32, WMMA).
//
// 3x3 convs as implicit GEMM: D[co,px] = sum_k W[co,k] * patch[k,px],
// k = ci*9 + kh*3 + kw.  bf16 in the MAC (v_wmma_f32_16x16x32_bf16, f32 acc),
// fp32 tensors in HBM.  Block tile: 16 co x 128 px, K staged 64-deep in LDS,
// 4 waves x (2 px-subtiles x 2 k-subchunks) = 4 WMMAs per wave per K-step.
// All spatial sizes are powers of two -> pure shift/mask addressing.
// ---------------------------------------------------------------------------

typedef __attribute__((ext_vector_type(16))) __bf16 v16bf;
typedef __attribute__((ext_vector_type(8)))  float  v8f;

union ABFrag { v16bf v; unsigned int u[8]; };
union CFrag  { v8f   v; float        f[8]; };

__device__ inline unsigned short f2bf(float f) {
    union { float f; unsigned int u; } x;
    x.f = f;
    unsigned int u = x.u;
    unsigned int r = u + 0x7FFFu + ((u >> 16) & 1u);   // round-to-nearest-even
    return (unsigned short)(r >> 16);
}

__device__ inline float silu_fast(float v) {
    // v * sigmoid(v) with hardware v_rcp_f32 (~1 ulp), no IEEE div sequence.
    return v * __builtin_amdgcn_rcpf(1.0f + __expf(-v));
}

#define CONV_THREADS 128   // 4 wave32s
#define KSTEP 64           // two 32-deep WMMA sub-chunks per LDS stage
#define PXT   128          // pixels per block

// Generic 3x3 conv, pad=1, stride 1 or 2, NCHW fp32, optional fused SiLU.
// grid.x = B*Hout*Wout/128 (pixel tiles), grid.y = Cout/16 (channel tiles).
// woutSh = log2(Wout), houtSh = log2(Hout)  (all sizes are powers of two).
__launch_bounds__(CONV_THREADS)
__global__ void conv3x3_silu_wmma(const float* __restrict__ in, long inBatchStride,
                                  const float* __restrict__ wgt,
                                  const float* __restrict__ bias,
                                  float* __restrict__ out,
                                  int Cin, int Hin, int Win,
                                  int Cout, int woutSh, int houtSh,
                                  int stride, int applySilu)
{
    // Row stride 68 shorts (136B) keeps k-pairs 4B-aligned, staggers banks.
    __shared__ unsigned short As[16][KSTEP + 4];    // [co][k]
    __shared__ unsigned short Bs[PXT][KSTEP + 4];   // [px][k]

    const int tid  = threadIdx.x;
    const int wave = tid >> 5;
    const int lane = tid & 31;
    const int m    = lane & 15;   // A row / B column within a 16x16 tile
    const int hh   = lane >> 4;   // half-wave select per ISA 7.12.2 layout

    const int  hwoSh  = woutSh + houtSh;        // log2(Hout*Wout)
    const int  Wout   = 1 << woutSh;
    const long pxBase = (long)blockIdx.x * PXT;
    const int  ct     = blockIdx.y;

    const int numK    = Cin * 9;
    const int nChunks = (numK + KSTEP - 1) / KSTEP;

    const float* wbase = wgt + (long)ct * 16 * numK;

    // ---- per-thread pixel column (thread == B-tile column), shift/mask ----
    const long pgT  = pxBase + tid;
    const int  bT   = (int)(pgT >> hwoSh);
    const int  pixT = (int)pgT & ((1 << hwoSh) - 1);
    const int  oyT  = pixT >> woutSh;
    const int  oxT  = pixT & (Wout - 1);
    const int  iy0  = oyT * stride - 1;     // +kh gives input row
    const int  ix0  = oxT * stride - 1;     // +kw gives input col
    const float* inbT = in + (long)bT * inBatchStride;

    CFrag acc0, acc1;
#pragma unroll
    for (int i = 0; i < 8; ++i) { acc0.f[i] = 0.0f; acc1.f[i] = 0.0f; }

    for (int kc = 0; kc < nChunks; ++kc) {
        const int k0 = kc * KSTEP;

        // ---- stage A: 16 co x 64 k weights -> bf16 LDS (coalesced in k) ----
#pragma unroll
        for (int it = 0; it < (16 * KSTEP) / CONV_THREADS; ++it) {
            int idx = tid + it * CONV_THREADS;
            int kk  = idx & (KSTEP - 1);
            int co  = idx >> 6;
            int k   = k0 + kk;
            int kl  = (k < numK) ? k : (numK - 1);        // clamp, no branch
            float v = wbase[(long)co * numK + kl];
            v = (k < numK) ? v : 0.0f;
            As[co][kk] = f2bf(v);
        }
        if (kc + 1 < nChunks && tid == 0)
            __builtin_prefetch(wbase + (long)(k0 + KSTEP), 0, 1);

        // ---- stage B: 64 k x 128 px patches, one pixel column per thread ----
        // (ci,kh,kw) decomposition of k is uniform across the wave -> SALU.
#pragma unroll 4
        for (int kk = 0; kk < KSTEP; ++kk) {
            int k  = k0 + kk;
            int kl = (k < numK) ? k : (numK - 1);
            int ci = kl / 9;
            int r  = kl - ci * 9;
            int kh = r / 3;
            int kw = r - kh * 3;
            int iy = iy0 + kh;
            int ix = ix0 + kw;
            bool valid = (k < numK) & (iy >= 0) & (iy < Hin) & (ix >= 0) & (ix < Win);
            int iyc = min(max(iy, 0), Hin - 1);           // clamp address,
            int ixc = min(max(ix, 0), Win - 1);           // select result
            float v = inbT[((long)ci * Hin + iyc) * Win + ixc];
            v = valid ? v : 0.0f;
            Bs[tid][kk] = f2bf(v);
        }
        __syncthreads();

        // ---- fragments per ISA 7.12.2 16-bit layouts ----
        // A 16x32: lane<16 row m holds K {0..7,16..23}; lane>=16 {8..15,24..31}
        // B 32x16: lane<16 col n holds K {0..7,16..23}; lane>=16 {8..15,24..31}
        ABFrag a0, a1, b00, b01, b10, b11;
        const int row0 = wave * 32 + m;        // px subtile 0
        const int row1 = wave * 32 + 16 + m;   // px subtile 1
#pragma unroll
        for (int i = 0; i < 8; ++i) {
            int ka = ((i < 4) ? (2 * i) : (2 * i + 8)) + 8 * hh;
            a0.u[i] = *(const unsigned int*)&As[m][ka];
            a1.u[i] = *(const unsigned int*)&As[m][32 + ka];
            int kb = ((i < 4) ? (2 * i) : (2 * i + 8)) + 8 * hh;
            b00.u[i] = *(const unsigned int*)&Bs[row0][kb];
            b01.u[i] = *(const unsigned int*)&Bs[row0][32 + kb];
            b10.u[i] = *(const unsigned int*)&Bs[row1][kb];
            b11.u[i] = *(const unsigned int*)&Bs[row1][32 + kb];
        }

        acc0.v = __builtin_amdgcn_wmma_f32_16x16x32_bf16(false, a0.v, false, b00.v,
                                                         (short)0, acc0.v, false, false);
        acc1.v = __builtin_amdgcn_wmma_f32_16x16x32_bf16(false, a0.v, false, b10.v,
                                                         (short)0, acc1.v, false, false);
        acc0.v = __builtin_amdgcn_wmma_f32_16x16x32_bf16(false, a1.v, false, b01.v,
                                                         (short)0, acc0.v, false, false);
        acc1.v = __builtin_amdgcn_wmma_f32_16x16x32_bf16(false, a1.v, false, b11.v,
                                                         (short)0, acc1.v, false, false);
        __syncthreads();
    }

    // ---- epilogue: D layout VGPR r -> (M = r + 8*hh, N = lane&15) ----
#pragma unroll
    for (int t = 0; t < 2; ++t) {
        const CFrag& acc = t ? acc1 : acc0;
        const int  pxl = wave * 32 + t * 16 + m;
        const long pg  = pxBase + pxl;
        const int  b   = (int)(pg >> hwoSh);
        const int  pix = (int)pg & ((1 << hwoSh) - 1);
#pragma unroll
        for (int r = 0; r < 8; ++r) {
            int   co = ct * 16 + r + 8 * hh;
            float v  = acc.f[r] + bias[co];
            if (applySilu) v = silu_fast(v);
            out[(((long)b * Cout + co) << hwoSh) + pix] = v;
        }
    }
}

// ---------------------------------------------------------------------------
// Forward softsplat (average mode): scatter-add 64 feature channels + a ones
// channel into acc[B][65][128][128] with bilinear corner weights.  H=W=128.
// ---------------------------------------------------------------------------
#define SPH 128
#define SPW 128
#define SPHW (SPH * SPW)

__global__ void softsplat_scatter(const float* __restrict__ feat,
                                  const float* __restrict__ flow, long flowBatchStride,
                                  float* __restrict__ acc, int C)
{
    long i  = (long)blockIdx.x * blockDim.x + threadIdx.x;
    int  b   = (int)(i >> 14);           // / 16384
    int  rem = (int)i & (SPHW - 1);
    int  y   = rem >> 7;
    int  x   = rem & (SPW - 1);

    const float* fl = flow + (long)b * flowBatchStride;
    float tx = (float)x + fl[rem];
    float ty = (float)y + fl[SPHW + rem];
    float x0 = floorf(tx), y0 = floorf(ty);
    float fx = tx - x0,    fy = ty - y0;
    int x0i = (int)x0, y0i = (int)y0;

    const float ws[4] = { (1.f - fx) * (1.f - fy), fx * (1.f - fy),
                          (1.f - fx) * fy,         fx * fy };
    const int  dxs[4] = { 0, 1, 0, 1 };
    const int  dys[4] = { 0, 0, 1, 1 };

    for (int c = 0; c <= C; ++c) {
        float val = (c < C) ? feat[(((long)b * C + c) << 14) + rem] : 1.0f;
        float* accC = acc + (((long)b * (C + 1) + c) << 14);
#pragma unroll
        for (int k = 0; k < 4; ++k) {
            int xi = x0i + dxs[k], yi = y0i + dys[k];
            if (xi >= 0 && xi < SPW && yi >= 0 && yi < SPH)
                atomicAdd(&accC[(yi << 7) + xi], val * ws[k]);
        }
    }
}

// xcat[b][c][y][x] : c<64 from acc_first, else acc_last; divide by ones-channel.
__global__ void normalize_concat(const float* __restrict__ accF,
                                 const float* __restrict__ accL,
                                 float* __restrict__ xcat, int C)
{
    long i   = (long)blockIdx.x * blockDim.x + threadIdx.x;
    int  b   = (int)(i >> 21);           // / (128 ch * 16384 px)
    int  r   = (int)i & ((128 << 14) - 1);
    int  c   = r >> 14;
    int  rem = r & (SPHW - 1);

    const float* a  = (c < C) ? accF : accL;
    int          cc = (c < C) ? c : c - C;
    float den = a[(((long)b * (C + 1) + C) << 14) + rem];
    den = (den == 0.0f) ? 1.0f : den;
    xcat[i] = a[(((long)b * (C + 1) + cc) << 14) + rem] / den;
}

__global__ void fillf(float* __restrict__ p, long n, float v)
{
    long i = (long)blockIdx.x * blockDim.x + threadIdx.x;
    if (i < n) p[i] = v;
}

// ---------------------------------------------------------------------------
// Host-side orchestration
// ---------------------------------------------------------------------------
extern "C" void kernel_launch(void* const* d_in, const int* in_sizes, int n_in,
                              void* d_out, int out_size, void* d_ws, size_t ws_size,
                              hipStream_t stream)
{
    (void)in_sizes; (void)n_in; (void)out_size; (void)ws_size;
    const int B = 8;

    // inputs (setup_inputs dict order)
    const float* lc   = (const float*)d_in[0];   // 8 x 6 x 512 x 512
    const float* flow = (const float*)d_in[1];   // 8 x 4 x 128 x 128
    const float* pfW[5]; const float* pfB[5];
    const float* plW[5]; const float* plB[5];
    for (int i = 0; i < 5; ++i) { pfW[i] = (const float*)d_in[2 + 2*i];  pfB[i] = (const float*)d_in[3 + 2*i]; }
    for (int i = 0; i < 5; ++i) { plW[i] = (const float*)d_in[12 + 2*i]; plB[i] = (const float*)d_in[13 + 2*i]; }
    const float* exW[4]; const float* exB[4];
    const float* zcW[4]; const float* zcB[4];
    for (int i = 0; i < 4; ++i) { exW[i] = (const float*)d_in[22 + 2*i]; exB[i] = (const float*)d_in[23 + 2*i]; }
    for (int i = 0; i < 4; ++i) { zcW[i] = (const float*)d_in[30 + 2*i]; zcB[i] = (const float*)d_in[31 + 2*i]; }

    // workspace bump allocator (floats), with aliasing after the pre stage
    float* ws = (float*)d_ws;
    size_t off = 0;
    auto alloc = [&](size_t n) { float* p = ws + off; off += n; return p; };
    float* bufA  = alloc((size_t)B * 16 * 512 * 512);   // 33.55 M floats
    float* bufB  = alloc((size_t)B * 32 * 256 * 256);   // 16.78 M
    float* bufC  = alloc((size_t)B * 32 * 256 * 256);   // 16.78 M
    float* tmp64 = alloc((size_t)B * 64 * 128 * 128);   //  8.39 M
    float* featF = alloc((size_t)B * 64 * 128 * 128);
    float* featL = alloc((size_t)B * 64 * 128 * 128);
    // aliases (pre-stage buffers are dead by the time these are live)
    const size_t accN = (size_t)B * 65 * 128 * 128;     // 8,519,680
    float* accF = bufA;
    float* accL = bufA + accN;
    float* xcat = bufB;                                  // exactly 16,777,216 floats
    float* x1   = bufC;
    float* x2   = bufC + (size_t)B * 192 * 64 * 64;
    float* x3   = x2   + (size_t)B * 256 * 32 * 32;

    float* out0 = (float*)d_out;
    float* out1 = out0 + (size_t)B * 192 * 64 * 64;   // 6,291,456
    float* out2 = out1 + (size_t)B * 256 * 32 * 32;   // +2,097,152
    float* out3 = out2 + (size_t)B * 384 * 16 * 16;   // +786,432

    // outSh is log2 of the (square) output spatial dim (all powers of two)
    auto conv = [&](const float* in, long ibs, const float* w, const float* bias, float* out,
                    int Cin, int Hin, int Win, int Cout, int outSh,
                    int stride, int silu) {
        long hwo = 1L << (2 * outSh);
        dim3 grid((unsigned)(((long)B * hwo) / PXT), (unsigned)(Cout / 16));
        conv3x3_silu_wmma<<<grid, CONV_THREADS, 0, stream>>>(
            in, ibs, w, bias, out, Cin, Hin, Win, Cout, outSh, outSh, stride, silu);
    };

    // ---- pre_first on channels 3:6, pre_last on channels 0:3 ----
    const long lcBS = 6L * 512 * 512;
    conv(lc + 3L * 512 * 512, lcBS, pfW[0], pfB[0], bufA, 3, 512, 512, 16, 9, 1, 1);
    conv(bufA, 16L * 512 * 512, pfW[1], pfB[1], bufB, 16, 512, 512, 32, 8, 2, 1);
    conv(bufB, 32L * 256 * 256, pfW[2], pfB[2], bufC, 32, 256, 256, 32, 8, 1, 1);
    conv(bufC, 32L * 256 * 256, pfW[3], pfB[3], tmp64, 32, 256, 256, 64, 7, 2, 1);
    conv(tmp64, 64L * 128 * 128, pfW[4], pfB[4], featF, 64, 128, 128, 64, 7, 1, 1);

    conv(lc, lcBS, plW[0], plB[0], bufA, 3, 512, 512, 16, 9, 1, 1);
    conv(bufA, 16L * 512 * 512, plW[1], plB[1], bufB, 16, 512, 512, 32, 8, 2, 1);
    conv(bufB, 32L * 256 * 256, plW[2], plB[2], bufC, 32, 256, 256, 32, 8, 1, 1);
    conv(bufC, 32L * 256 * 256, plW[3], plB[3], tmp64, 32, 256, 256, 64, 7, 2, 1);
    conv(tmp64, 64L * 128 * 128, plW[4], plB[4], featL, 64, 128, 128, 64, 7, 1, 1);

    // ---- softsplat-average both directions ----
    {
        long n = 2L * (long)accN;
        fillf<<<dim3((unsigned)((n + 255) / 256)), 256, 0, stream>>>(accF, n, 0.0f);
        long px = (long)B * 128 * 128;                  // multiple of 256
        dim3 g((unsigned)(px / 256));
        const long flBS = 4L * 128 * 128;
        softsplat_scatter<<<g, 256, 0, stream>>>(featF, flow,                  flBS, accF, 64);
        softsplat_scatter<<<g, 256, 0, stream>>>(featL, flow + 2L * 128 * 128, flBS, accL, 64);
        long nc = (long)B * 128 * 128 * 128;            // multiple of 256
        normalize_concat<<<dim3((unsigned)(nc / 256)), 256, 0, stream>>>(accF, accL, xcat, 64);
    }

    // ---- extractor pyramid + zero convs -> d_out ----
    conv(xcat, 128L * 128 * 128, exW[0], exB[0], x1, 128, 128, 128, 192, 6, 2, 1);
    conv(x1, 192L * 64 * 64,     zcW[0], zcB[0], out0, 192, 64, 64, 192, 6, 1, 0);
    conv(x1, 192L * 64 * 64,     exW[1], exB[1], x2, 192, 64, 64, 256, 5, 2, 1);
    conv(x2, 256L * 32 * 32,     zcW[1], zcB[1], out1, 256, 32, 32, 256, 5, 1, 0);
    conv(x2, 256L * 32 * 32,     exW[2], exB[2], x3, 256, 32, 32, 384, 4, 2, 1);
    conv(x3, 384L * 16 * 16,     zcW[2], zcB[2], out2, 384, 16, 16, 384, 4, 1, 0);
    conv(x3, 384L * 16 * 16,     exW[3], exB[3], x2, 384, 16, 16, 512, 3, 2, 1);  // reuse x2 slot
    conv(x2, 512L * 8 * 8,       zcW[3], zcB[3], out3, 512, 8, 8, 512, 3, 1, 0);
}
